// SSKernel_37993280700536
// MI455X (gfx1250) — compile-verified, matrix operand order checked by reading, not run on hardware
//
#include <hip/hip_runtime.h>

typedef __attribute__((ext_vector_type(16))) _Float16 v16h;
typedef __attribute__((ext_vector_type(8)))  float    v8f;

#define HDIM   512
#define NDIM   64
#define LDIM   4096
#define NWAVES 8
#define TILE_L 256   // 16 lhi * 16 llo outputs per WMMA tile

// Pack two f32 -> adjacent f16 halves, single 32-bit LDS store.
__device__ __forceinline__ void store_h2(_Float16* p, float a, float b) {
    union { _Float16 h[2]; unsigned u; } x;
    x.h[0] = (_Float16)a;
    x.h[1] = (_Float16)b;
    *(unsigned*)p = x.u;
}

// K[h, l0+16*lhi+llo] = 2 * Re( sum_n W[lhi][n] * Q[n][llo] )
//   W[lhi][n] = coef[n] * exp(dtA[n]*(l0+16*lhi))  -> recurrence from exact base
//   Q[n][llo] = exp(dtA[n]*llo)                    -> shared across block
// Real contraction K=128: A[k=2n]=Wr, A[2n+1]=-Wi ; B[2n]=Qr, B[2n+1]=Qi.
__global__ __launch_bounds__(256)
void s4d_wmma_kernel(const float* __restrict__ g_logdt,
                     const float* __restrict__ g_C,
                     const float* __restrict__ g_B,
                     const float* __restrict__ g_invAr,
                     const float* __restrict__ g_Aimag,
                     float* __restrict__ g_out)
{
    __shared__ _Float16 Ash[NWAVES][16][128];  // per-wave W tile, row-major (32 KB)
    __shared__ _Float16 Qsh[16][128];          // Q, llo-major for contiguous B frags (4 KB)

    const int h    = blockIdx.x;
    const int tid  = threadIdx.x;
    const int wave = tid >> 5;
    const int lane = tid & 31;            // wave32

    const float dt = __expf(g_logdt[h]);

    // ---- Shared Q fill: 64 n * 16 llo complex entries, 4 per thread.
    for (int i = 0; i < 4; ++i) {
        int idx = tid + 256 * i;          // 0..1023
        int n   = idx >> 4;
        int llo = idx & 15;
        float Ar = -__expf(g_invAr[h * NDIM + n]);
        float Ai =  g_Aimag[h * NDIM + n];
        float dtAr = Ar * dt, dtAi = Ai * dt;
        float ex = __expf(dtAr * (float)llo);
        float qr = ex * __cosf(dtAi * (float)llo);
        float qi = ex * __sinf(dtAi * (float)llo);
        store_h2(&Qsh[llo][2 * n], qr, qi);
    }

    // ---- Per-lane ZOH coefficients + 16-step phase factor for n = lane, lane+32.
    float coefr[2], coefi[2], dtAr_[2], dtAi_[2], s16r[2], s16i[2];
    #pragma unroll
    for (int j = 0; j < 2; ++j) {
        int n = lane + 32 * j;
        float Ar = -__expf(g_invAr[h * NDIM + n]);
        float Ai =  g_Aimag[h * NDIM + n];
        float dtAr = Ar * dt, dtAi = Ai * dt;
        // exp(dtA) - 1
        float er = __expf(dtAr);
        float zr = er * __cosf(dtAi) - 1.0f;
        float zi = er * __sinf(dtAi);
        // Bc * Cc
        int ci = (h * NDIM + n) * 2;
        float cr = g_C[ci], cim = g_C[ci + 1];
        float br = g_B[ci], bi  = g_B[ci + 1];
        float tr = cr * br - cim * bi;
        float ti = cr * bi + cim * br;
        // * (exp(dtA)-1)
        float numr = tr * zr - ti * zi;
        float numi = tr * zi + ti * zr;
        // / A
        float inv = 1.0f / (Ar * Ar + Ai * Ai);
        coefr[j] = (numr * Ar + numi * Ai) * inv;
        coefi[j] = (numi * Ar - numr * Ai) * inv;
        dtAr_[j] = dtAr;
        dtAi_[j] = dtAi;
        // step16 = exp(dtA * 16)
        float e16 = __expf(dtAr * 16.0f);
        s16r[j] = e16 * __cosf(dtAi * 16.0f);
        s16i[j] = e16 * __sinf(dtAi * 16.0f);
    }

    const int m     = lane & 15;              // A-frag row index
    const int nn    = lane & 15;              // B-frag col / D column index
    const int hiSel = (lane >= 16) ? 8 : 0;   // 16-bit frag K-offset / D M-offset

    for (int rep = 0; rep < 2; ++rep) {
        const int l0 = (wave + NWAVES * rep) * TILE_L;

        // ---- Build this wave's W tile: exact base at l0, then x step16 recurrence.
        #pragma unroll
        for (int j = 0; j < 2; ++j) {
            int n = lane + 32 * j;
            float lf = (float)l0;
            float ex = __expf(dtAr_[j] * lf);
            float pr = ex * __cosf(dtAi_[j] * lf);
            float pi = ex * __sinf(dtAi_[j] * lf);
            float wr = coefr[j] * pr - coefi[j] * pi;
            float wi = coefr[j] * pi + coefi[j] * pr;
            #pragma unroll
            for (int lhi = 0; lhi < 16; ++lhi) {
                store_h2(&Ash[wave][lhi][2 * n], wr, -wi);
                float nwr = wr * s16r[j] - wi * s16i[j];
                float nwi = wr * s16i[j] + wi * s16r[j];
                wr = nwr; wi = nwi;
            }
        }
        __syncthreads();

        // ---- K=128 contraction: 4 x v_wmma_f32_16x16x32_f16, f32 accumulate.
        v8f acc = {};
        #pragma unroll
        for (int c = 0; c < 4; ++c) {
            v16h av, bv;
            // 16-bit A/B fragment layout (ISA 7.12.2):
            //   half t -> K = 32c + t + 8*(t>=8) + 8*(lane>=16)
            // Both Ash rows and Qsh rows are contiguous in K -> 2x ds_load_b128 each.
            #pragma unroll
            for (int t = 0; t < 16; ++t) {
                int k = 32 * c + t + ((t >= 8) ? 8 : 0) + hiSel;
                av[t] = Ash[wave][m][k];
                bv[t] = Qsh[nn][k];
            }
            acc = __builtin_amdgcn_wmma_f32_16x16x32_f16(
                /*neg_a=*/false, av, /*neg_b=*/false, bv,
                /*c_mod=*/(short)0, acc, /*reuse_a=*/false, /*reuse_b=*/false);
        }

        // ---- D layout: VGPR v -> M = v + hiSel (lhi), N = lane&15 (llo).
        float* outp = g_out + h * LDIM + l0;
        #pragma unroll
        for (int v = 0; v < 8; ++v)
            outp[16 * (v + hiSel) + nn] = 2.0f * acc[v];

        __syncthreads();
    }
}

extern "C" void kernel_launch(void* const* d_in, const int* in_sizes, int n_in,
                              void* d_out, int out_size, void* d_ws, size_t ws_size,
                              hipStream_t stream) {
    const float* g_logdt = (const float*)d_in[0];  // (H,)
    const float* g_C     = (const float*)d_in[1];  // (1, H, N, 2)
    const float* g_B     = (const float*)d_in[2];  // (S=H, N, 2)
    const float* g_invAr = (const float*)d_in[3];  // (S=H, N)
    const float* g_Aimag = (const float*)d_in[4];  // (S=H, N)
    // d_in[5] = L (scalar, fixed at 4096)
    float* g_out = (float*)d_out;                  // (1, H, L)

    s4d_wmma_kernel<<<dim3(HDIM), dim3(256), 0, stream>>>(
        g_logdt, g_C, g_B, g_invAr, g_Aimag, g_out);
}
